// MemoryEfficientAttention_1692217114780
// MI455X (gfx1250) — matrix operand order repaired and checked by reading, not compiled, not run on hardware
//
#include <hip/hip_runtime.h>

typedef _Float16 half_t;
typedef __attribute__((ext_vector_type(16))) _Float16 v16h;
typedef __attribute__((ext_vector_type(8)))  _Float16 v8h;
typedef __attribute__((ext_vector_type(4)))  _Float16 v4h;
typedef __attribute__((ext_vector_type(8)))  float    v8f;
typedef __attribute__((ext_vector_type(4)))  float    v4f;

#define S_LEN 1024
#define D_DIM 64
#define BR    128     // query rows per block (8 waves x 16)
#define BC    64      // key rows per iteration
#define NIT   (S_LEN / BC)
#define LDK   72      // LDS row stride in halves (144B: 16B-aligned, bank-skewed)
#define LDP   72

__device__ __forceinline__ v16h concat8(v8h a, v8h b) {
    return __builtin_shufflevector(a, b, 0,1,2,3,4,5,6,7,8,9,10,11,12,13,14,15);
}

// ---- DPP row-rotate (within 16-lane rows) butterfly reductions ------------
// row_ror:n -> dpp_ctrl = 0x120 + n ; never crosses the 16-lane half-wave.
template <int C>
__device__ __forceinline__ float dpp_mov(float x) {
    return __builtin_bit_cast(float,
        __builtin_amdgcn_update_dpp(0, __builtin_bit_cast(int, x),
                                    C, 0xf, 0xf, true));
}
__device__ __forceinline__ float row16_max(float x) {
    x = fmaxf(x, dpp_mov<0x121>(x));   // ror 1
    x = fmaxf(x, dpp_mov<0x122>(x));   // ror 2
    x = fmaxf(x, dpp_mov<0x124>(x));   // ror 4
    x = fmaxf(x, dpp_mov<0x128>(x));   // ror 8
    return x;
}
__device__ __forceinline__ float row16_sum(float x) {
    x += dpp_mov<0x121>(x);
    x += dpp_mov<0x122>(x);
    x += dpp_mov<0x124>(x);
    x += dpp_mov<0x128>(x);
    return x;
}

// ---- tile pipeline helpers ------------------------------------------------
__device__ __forceinline__ void load_tile(const float* kb, const float* vb,
                                          int kb0, int tid, v4f* kr, v4f* vr) {
#pragma unroll
    for (int j = 0; j < 4; ++j) {
        const int e   = 4 * (tid + 256 * j);
        const int row = e >> 6;
        const int col = e & 63;
        kr[j] = *(const v4f*)(kb + (size_t)(kb0 + row) * D_DIM + col);
        vr[j] = *(const v4f*)(vb + (size_t)(kb0 + row) * D_DIM + col);
    }
}

__device__ __forceinline__ void stage_tile(half_t* sKb, half_t* sVtb,
                                           const v4f* kr, const v4f* vr, int tid) {
#pragma unroll
    for (int j = 0; j < 4; ++j) {
        const int e   = 4 * (tid + 256 * j);
        const int row = e >> 6;
        const int col = e & 63;
        v4h kh;
        kh[0] = (half_t)kr[j][0]; kh[1] = (half_t)kr[j][1];
        kh[2] = (half_t)kr[j][2]; kh[3] = (half_t)kr[j][3];
        *(v4h*)&sKb[row * LDK + col] = kh;           // K row-major
        sVtb[(col + 0) * LDK + row] = (half_t)vr[j][0];  // V transposed
        sVtb[(col + 1) * LDK + row] = (half_t)vr[j][1];
        sVtb[(col + 2) * LDK + row] = (half_t)vr[j][2];
        sVtb[(col + 3) * LDK + row] = (half_t)vr[j][3];
    }
}

__global__ __launch_bounds__(256, 1)
void fa_fwd_kernel(const float* __restrict__ q,
                   const float* __restrict__ k,
                   const float* __restrict__ v,
                   float* __restrict__ out)
{
    __shared__ half_t sK [2][BC * LDK];      // double-buffered K tile [key][d]
    __shared__ half_t sVt[2][D_DIM * LDK];   // double-buffered V^T    [d][key]
    __shared__ half_t sP [8][16 * LDP];      // per-wave P tile        [row][key]

    const int tid  = threadIdx.x;
    const int lane = tid & 31;
    const int wave = tid >> 5;
    const int lh   = lane >> 4;   // half-wave: 0 or 1
    const int lm   = lane & 15;   // M (A rows) or N (B cols) index

    const int b     = blockIdx.y;
    const int qbase = blockIdx.x * BR + wave * 16;

    const float* qb = q + ((size_t)b * S_LEN + qbase) * D_DIM;
    const float* kb = k + (size_t)b * S_LEN * D_DIM;
    const float* vb = v + (size_t)b * S_LEN * D_DIM;

    // ---- Q A-fragments: 16x32 f16 each (ISA 7.12.2 A layout) --------------
    v16h qf[2];
#pragma unroll
    for (int f = 0; f < 2; ++f) {
        const int kA0 = f * 32 + lh * 8;
        const float* rp = qb + lm * D_DIM;
        v4f t0a = *(const v4f*)(rp + kA0);
        v4f t0b = *(const v4f*)(rp + kA0 + 4);
        v4f t1a = *(const v4f*)(rp + kA0 + 16);
        v4f t1b = *(const v4f*)(rp + kA0 + 20);
        v8h lo, hi;
#pragma unroll
        for (int i = 0; i < 4; ++i) {
            lo[i]     = (half_t)t0a[i];
            lo[i + 4] = (half_t)t0b[i];
            hi[i]     = (half_t)t1a[i];
            hi[i + 4] = (half_t)t1b[i];
        }
        qf[f] = concat8(lo, hi);
    }

    // ---- accumulator state ------------------------------------------------
    v8f o[4];
    float m[8], l[8];
    {
        v8f z = {};
#pragma unroll
        for (int n = 0; n < 4; ++n) o[n] = z;
#pragma unroll
        for (int r = 0; r < 8; ++r) { m[r] = -3.0e38f; l[r] = 0.0f; }
    }

    // ---- pipeline prologue: stage tile 0 ----------------------------------
    v4f kr[4], vr[4];
    load_tile(kb, vb, 0, tid, kr, vr);
    stage_tile(&sK[0][0], &sVt[0][0], kr, vr, tid);
    __syncthreads();

    // ---- main loop: double-buffered, one barrier per iteration ------------
    int p = 0;
    for (int it = 0; it < NIT; ++it, p ^= 1) {
        if (it + 1 < NIT) {
            load_tile(kb, vb, (it + 1) * BC, tid, kr, vr);   // in-flight during compute
            if (it + 2 < NIT) {                              // HBM -> L2 for tile i+2
                __builtin_prefetch(kb + (size_t)(it + 2) * BC * D_DIM + tid * 16, 0, 3);
                __builtin_prefetch(vb + (size_t)(it + 2) * BC * D_DIM + tid * 16, 0, 3);
            }
        }

        const half_t* sKp  = &sK [p][0];
        const half_t* sVtp = &sVt[p][0];

        // S = Q * K^T : 4 tiles of 16x16, each 2 WMMAs (K=32)
        v8f sc[4];
#pragma unroll
        for (int n = 0; n < 4; ++n) {
            v8f acc = {};
#pragma unroll
            for (int f = 0; f < 2; ++f) {
                const half_t* bp = &sKp[(n * 16 + lm) * LDK + f * 32 + lh * 8];
                v16h bf = concat8(*(const v8h*)bp, *(const v8h*)(bp + 16));
                acc = __builtin_amdgcn_wmma_f32_16x16x32_f16(
                        false, qf[f], false, bf, (short)0, acc, false, false);
            }
            sc[n] = acc;
        }

        // online softmax (rows live in 16-lane halves; DPP reductions)
#pragma unroll
        for (int r = 0; r < 8; ++r) {
            float mx = fmaxf(fmaxf(sc[0][r], sc[1][r]), fmaxf(sc[2][r], sc[3][r]));
            mx = row16_max(mx);
            const float mnew = fmaxf(m[r], mx);
            const float corr = __expf(m[r] - mnew);
            m[r] = mnew;
            float rs = 0.0f;
#pragma unroll
            for (int n = 0; n < 4; ++n) {
                float pj = __expf(sc[n][r] - mnew);
                sc[n][r] = pj;
                rs += pj;
            }
            rs = row16_sum(rs);
            l[r] = l[r] * corr + rs;
#pragma unroll
            for (int n = 0; n < 4; ++n) o[n][r] *= corr;
        }

        // stage P as f16 into this wave's LDS region (same-wave DScnt RAW)
        half_t* pw = &sP[wave][0];
#pragma unroll
        for (int n = 0; n < 4; ++n)
#pragma unroll
            for (int r = 0; r < 8; ++r)
                pw[(r + lh * 8) * LDP + n * 16 + lm] = (half_t)sc[n][r];

        // O += P * V : A = P (16x32), B = V^T (32x16)
#pragma unroll
        for (int s0 = 0; s0 < 2; ++s0) {
            const int kA0 = s0 * 32 + lh * 8;
            const half_t* ap = pw + lm * LDP + kA0;
            v16h af = concat8(*(const v8h*)ap, *(const v8h*)(ap + 16));
#pragma unroll
            for (int n = 0; n < 4; ++n) {
                const half_t* bp = &sVtp[(n * 16 + lm) * LDK + s0 * 32 + lh * 8];
                v16h bf = concat8(*(const v8h*)bp, *(const v8h*)(bp + 16));
                o[n] = __builtin_amdgcn_wmma_f32_16x16x32_f16(
                         false, af, false, bf, (short)0, o[n], false, false);
            }
        }

        if (it + 1 < NIT)
            stage_tile(&sK[p ^ 1][0], &sVt[p ^ 1][0], kr, vr, tid);
        __syncthreads();
    }

    // ---- epilogue: normalize and store ------------------------------------
    float* ob = out + ((size_t)b * S_LEN + qbase) * D_DIM;
#pragma unroll
    for (int r = 0; r < 8; ++r) {
        const float inv = 1.0f / l[r];
        const int   row = r + lh * 8;
#pragma unroll
        for (int n = 0; n < 4; ++n)
            ob[row * D_DIM + n * 16 + lm] = o[n][r] * inv;
    }
}

extern "C" void kernel_launch(void* const* d_in, const int* in_sizes, int n_in,
                              void* d_out, int out_size, void* d_ws, size_t ws_size,
                              hipStream_t stream) {
    const float* q = (const float*)d_in[0];
    const float* k = (const float*)d_in[1];
    const float* v = (const float*)d_in[2];
    float* out = (float*)d_out;

    const int B = in_sizes[0] / (S_LEN * D_DIM);   // 128
    dim3 grid(S_LEN / BR, B);                      // (8, 128)
    dim3 block(256);                               // 8 wave32 waves
    fa_fwd_kernel<<<grid, block, 0, stream>>>(q, k, v, out);
}
